// TiSASRec_56530359550515
// MI455X (gfx1250) — compile-verified, matrix operand order checked by reading, not run on hardware
//
#include <hip/hip_runtime.h>
#include <hip/hip_bf16.h>

// ---------------- problem constants ----------------
#define Bc   32
#define Lq   200
#define Lp   224          // padded to 14*16 = 7*32
#define Dm   64
#define DH   32           // head dim
#define NH   2
#define NLAY 2
#define TSP  366          // time buckets 0..365
#define TPp  384          // padded buckets (24*16, 12*32)
#define NROWS (Bc*Lq)     // 6400

__constant__ float kNEG = -4294967295.0f;      // -2^32+1
#define ISQ 0.17677669529663687f               // 1/sqrt(32)

// ---------------- WMMA vector types ----------------
typedef __attribute__((ext_vector_type(16))) _Float16 v16h;
typedef __attribute__((ext_vector_type(8)))  _Float16 h8;
typedef __attribute__((ext_vector_type(8)))  float    v8f;

__device__ __forceinline__ v16h mk16(h8 lo, h8 hi) {
    v16h r;
#pragma unroll
    for (int i = 0; i < 8; ++i) { r[i] = lo[i]; r[i + 8] = hi[i]; }
    return r;
}
// 16-bit A/B tile per-lane layout: element i -> col K(i) = (i<8 ? i : i+8) + 8*hl
// => two contiguous 8-half runs at kb+8*hl and kb+16+8*hl.
__device__ __forceinline__ v16h load_row_f16(const _Float16* row, int kb, int hl) {
    h8 lo = *(const h8*)(row + kb + 8 * hl);
    h8 hi = *(const h8*)(row + kb + 16 + 8 * hl);
    return mk16(lo, hi);
}
__device__ __forceinline__ h8 cv8(const float* p) {
    h8 r;
#pragma unroll
    for (int i = 0; i < 8; ++i) r[i] = (_Float16)p[i];
    return r;
}
__device__ __forceinline__ v16h load_row_f32cvt(const float* row, int kb, int hl) {
    return mk16(cv8(row + kb + 8 * hl), cv8(row + kb + 16 + 8 * hl));
}
__device__ __forceinline__ v16h zero16() {
    v16h r;
#pragma unroll
    for (int i = 0; i < 16; ++i) r[i] = (_Float16)0.0f;
    return r;
}
__device__ __forceinline__ v8f zero8() {
    v8f r;
#pragma unroll
    for (int i = 0; i < 8; ++i) r[i] = 0.0f;
    return r;
}
#define WMMA_F16(A, B, C) __builtin_amdgcn_wmma_f32_16x16x32_f16(false, (A), false, (B), (short)0, (C), false, false)

// ---------------- kernels ----------------

// x[r,d] = item_emb[ids[r]][d] * 8 * (id != 0)
__global__ void embed_kernel(const int* __restrict__ ids, const float* __restrict__ emb,
                             float* __restrict__ x) {
    int r = blockIdx.x, d = threadIdx.x;
    int id = ids[r];
    float v = emb[(size_t)id * Dm + d] * 8.0f;
    x[(size_t)r * Dm + d] = (id == 0) ? 0.0f : v;
}

// LayerNorm over 64-wide rows with optional fused residual add:
// dst = LN(src + add) ; one wave32 per row (2 elems/lane), full shfl reduce.
__global__ void ln_kernel(const float* __restrict__ src, const float* __restrict__ add,
                          float* __restrict__ dst, const float* __restrict__ g,
                          const float* __restrict__ bb, int nrows) {
    int row = blockIdx.x * 8 + (threadIdx.x >> 5);
    int lane = threadIdx.x & 31;
    if (row >= nrows) return;
    const float* p = src + (size_t)row * Dm;
    float e0 = p[lane], e1 = p[lane + 32];
    if (add) {
        const float* a = add + (size_t)row * Dm;
        e0 += a[lane];
        e1 += a[lane + 32];
    }
    float s = e0 + e1;
#pragma unroll
    for (int m = 1; m < 32; m <<= 1) s += __shfl_xor(s, m, 32);
    float mean = s * (1.0f / 64.0f);
    float d0 = e0 - mean, d1 = e1 - mean;
    float v = d0 * d0 + d1 * d1;
#pragma unroll
    for (int m = 1; m < 32; m <<= 1) v += __shfl_xor(v, m, 32);
    float inv = rsqrtf(v * (1.0f / 64.0f) + 1e-8f);
    float* q = dst + (size_t)row * Dm;
    q[lane]      = d0 * inv * g[lane]      + bb[lane];
    q[lane + 32] = d1 * inv * g[lane + 32] + bb[lane + 32];
}

// out_f16[b,l(pad Lp),n] = relu?( src[b,l,:] @ W[n,:] + bias[n] + posadd[l,n] )
// WMMA GEMM: blockDim=128 (4 waves, one 16-col N-tile each); grid = B*Lp/16 M-tiles.
__global__ void proj_kernel(const float* __restrict__ src, const float* __restrict__ W,
                            const float* __restrict__ bias, const float* __restrict__ posadd,
                            int relu, _Float16* __restrict__ out) {
    int mt = blockIdx.x;
    int wv = threadIdx.x >> 5;
    int lane = threadIdx.x & 31;
    int nn = lane & 15, hl = lane >> 4;
    int b = (mt * 16) / Lp;
    int l0 = (mt * 16) % Lp;
    int n0 = wv * 16;

    v8f c = zero8();
#pragma unroll
    for (int ks = 0; ks < 2; ++ks) {
        int l = l0 + nn;
        v16h a = (l < Lq) ? load_row_f32cvt(src + ((size_t)b * Lq + l) * Dm, ks * 32, hl)
                          : zero16();
        v16h bw = load_row_f32cvt(W + (size_t)(n0 + nn) * Dm, ks * 32, hl);
        c = WMMA_F16(a, bw, c);
    }
#pragma unroll
    for (int j = 0; j < 8; ++j) {
        int l = l0 + j + 8 * hl;
        if (l < Lq) {
            int n = n0 + nn;
            float v = c[j] + bias[n];
            if (posadd) v += posadd[(size_t)l * Dm + n];
            if (relu) v = fmaxf(v, 0.0f);
            out[((size_t)b * Lp + l) * Dm + n] = (_Float16)v;
        }
    }
}

// QT[b,h,q,t] = Qh[b,q,h*32:]. tKe[t,h*32:]   (WMMA, K=32 single step)
__global__ void qt_kernel(const _Float16* __restrict__ Qh, const float* __restrict__ tKe,
                          _Float16* __restrict__ QT) {
    int gx = blockIdx.x;                 // b*NH*14
    int mt = gx % 14;
    int h = (gx / 14) % NH;
    int b = gx / (14 * NH);
    int t0 = blockIdx.y * 16;
    int lane = threadIdx.x;
    int nn = lane & 15, hl = lane >> 4;
    int q0 = mt * 16;

    v16h a = load_row_f16(Qh + ((size_t)b * Lp + q0 + nn) * Dm + h * DH, 0, hl);
    int t = t0 + nn;
    v16h bt = (t < TSP) ? load_row_f32cvt(tKe + (size_t)t * Dm + h * DH, 0, hl) : zero16();
    v8f c = zero8();
    c = WMMA_F16(a, bt, c);
    _Float16* QTb = QT + (((size_t)b * NH + h) * Lp) * TPp;
#pragma unroll
    for (int j = 0; j < 8; ++j)
        QTb[(size_t)(q0 + j + 8 * hl) * TPp + t0 + nn] = (_Float16)c[j];
}

// Fused attention per (b, h, q-tile): single wave32 workgroup.
// score QK't  : WMMA (14 tiles, K-biases from global, L2-hot)
// softmax     : in-register, 16-lane shfl reductions
// out A@V'    : WMMA (7 k-steps x 2 N-tiles), V' transposed in LDS
// time-value  : bucket scatter-add (ds_add_f32) -> W[16,384], then WMMA W@tVe
__global__ __launch_bounds__(32) void attn_kernel(
        const _Float16* __restrict__ Qh, const _Float16* __restrict__ Kp,
        const _Float16* __restrict__ Vp, const _Float16* __restrict__ QT,
        const int* __restrict__ ltimes, const int* __restrict__ lids,
        const float* __restrict__ tVe, float* __restrict__ out) {
    __shared__ _Float16 VpT_s[DH * Lp];      // [d][k] transposed V'
    __shared__ _Float16 tVeT_s[DH * TPp];    // [d][t] transposed time-V table slice
    __shared__ _Float16 probs_s[16 * Lp];    // softmax probs, row-major
    __shared__ float    W_s[16 * TPp];       // per-bucket prob mass
    __shared__ int times_s[Lp];
    __shared__ int pad_s[Lp];

    int bid = blockIdx.x;
    int qt = bid % 14;
    int h = (bid / 14) % NH;
    int b = bid / (14 * NH);
    int lane = threadIdx.x;
    int nn = lane & 15, hl = lane >> 4;
    int q0 = qt * 16;

    const _Float16* QTb = QT + (((size_t)b * NH + h) * Lp) * TPp;
    __builtin_prefetch(QTb + (size_t)q0 * TPp, 0, 0);          // global_prefetch
    __builtin_prefetch(QTb + (size_t)(q0 + 8) * TPp, 0, 0);

    for (int l = lane; l < Lp; l += 32) {
        times_s[l] = (l < Lq) ? ltimes[b * Lq + l] : 0;
        pad_s[l]   = (l < Lq) ? (lids[b * Lq + l] == 0) : 1;
    }
    for (int i = lane; i < Lp * DH; i += 32) {
        int k = i >> 5, d = i & 31;
        VpT_s[d * Lp + k] = Vp[((size_t)b * Lp + k) * Dm + h * DH + d];
    }
    for (int i = lane; i < DH * TPp; i += 32) {
        int d = i / TPp, t = i % TPp;
        tVeT_s[i] = (t < TSP) ? (_Float16)tVe[(size_t)t * Dm + h * DH + d] : (_Float16)0.0f;
    }
    for (int i = lane; i < 16 * TPp; i += 32) W_s[i] = 0.0f;
    __syncthreads();

    // ---- scores: Q @ K'^T (K' = K + posK folded upstream; B-tiles from global) ----
    v16h aq = load_row_f16(Qh + ((size_t)b * Lp + q0 + nn) * Dm + h * DH, 0, hl);
    v8f c[14];
#pragma unroll
    for (int kt = 0; kt < 14; ++kt) {
        v16h bk = load_row_f16(Kp + ((size_t)b * Lp + kt * 16 + nn) * Dm + h * DH, 0, hl);
        v8f acc = zero8();
        c[kt] = WMMA_F16(aq, bk, acc);
    }

    // ---- + time-key term (gather from QT), scale, mask ----
    int tq[8];
#pragma unroll
    for (int j = 0; j < 8; ++j) tq[j] = times_s[q0 + j + 8 * hl];
#pragma unroll
    for (int kt = 0; kt < 14; ++kt) {
        int k = kt * 16 + nn;
        int tk = times_s[k];
#pragma unroll
        for (int j = 0; j < 8; ++j) {
            int q = q0 + j + 8 * hl;
            int tm = (int)fminf(fabsf((float)(tq[j] - tk)) * (1.0f / 86400.0f), 365.0f);
            float s = (c[kt][j] + (float)QTb[(size_t)q * TPp + tm]) * ISQ;
            bool msk = (k > q) || (k >= Lq) || pad_s[q];
            c[kt][j] = msk ? kNEG : s;
        }
    }

    // ---- softmax over 224 cols, rows distributed 16-lanes wide ----
#pragma unroll
    for (int j = 0; j < 8; ++j) {
        float m = c[0][j];
#pragma unroll
        for (int kt = 1; kt < 14; ++kt) m = fmaxf(m, c[kt][j]);
#pragma unroll
        for (int msk = 1; msk < 16; msk <<= 1) m = fmaxf(m, __shfl_xor(m, msk, 32));
        float ssum = 0.0f;
#pragma unroll
        for (int kt = 0; kt < 14; ++kt) {
            float e = __expf(c[kt][j] - m);
            c[kt][j] = e;
            ssum += e;
        }
#pragma unroll
        for (int msk = 1; msk < 16; msk <<= 1) ssum += __shfl_xor(ssum, msk, 32);
        float rinv = 1.0f / ssum;
#pragma unroll
        for (int kt = 0; kt < 14; ++kt) c[kt][j] *= rinv;
    }
#pragma unroll
    for (int kt = 0; kt < 14; ++kt)
#pragma unroll
        for (int j = 0; j < 8; ++j)
            probs_s[(j + 8 * hl) * Lp + kt * 16 + nn] = (_Float16)c[kt][j];
    __syncthreads();

    // ---- time-value bucket scatter: W[m,t] += A[m,k] for tmat(q,k)==t ----
#pragma unroll 1
    for (int m = 0; m < 16; ++m) {
        int tqm = times_s[q0 + m];
#pragma unroll 1
        for (int k = lane; k < Lq; k += 32) {
            float p = (float)probs_s[m * Lp + k];
            if (p != 0.0f) {
                int tm = (int)fminf(fabsf((float)(tqm - times_s[k])) * (1.0f / 86400.0f), 365.0f);
                atomicAdd(&W_s[m * TPp + tm], p);     // ds_add_f32
            }
        }
    }

    // ---- out = A @ V' (WMMA over 7 k-steps, 2 N-tiles for dh=32) ----
    v8f o0 = zero8(), o1 = zero8();
#pragma unroll
    for (int ks = 0; ks < 7; ++ks) {
        v16h ap = load_row_f16(&probs_s[nn * Lp], ks * 32, hl);
        v16h b0 = load_row_f16(&VpT_s[(0 + nn) * Lp], ks * 32, hl);
        v16h b1 = load_row_f16(&VpT_s[(16 + nn) * Lp], ks * 32, hl);
        o0 = WMMA_F16(ap, b0, o0);
        o1 = WMMA_F16(ap, b1, o1);
    }
    __syncthreads();

    // ---- + W @ tVe (WMMA over 12 k-steps of 32 buckets) ----
#pragma unroll
    for (int ks = 0; ks < 12; ++ks) {
        v16h aw = load_row_f32cvt(&W_s[nn * TPp], ks * 32, hl);
        v16h b0 = load_row_f16(&tVeT_s[(0 + nn) * TPp], ks * 32, hl);
        v16h b1 = load_row_f16(&tVeT_s[(16 + nn) * TPp], ks * 32, hl);
        o0 = WMMA_F16(aw, b0, o0);
        o1 = WMMA_F16(aw, b1, o1);
    }

    // ---- store ----
#pragma unroll
    for (int j = 0; j < 8; ++j) {
        int q = q0 + j + 8 * hl;
        if (q < Lq) {
            out[((size_t)b * Lq + q) * Dm + h * DH + nn]      = o0[j];
            out[((size_t)b * Lq + q) * Dm + h * DH + 16 + nn] = o1[j];
        }
    }
}

// x = keep * (x + Ht @ W2^T + b2)
__global__ void ffn2_kernel(const _Float16* __restrict__ Ht, const float* __restrict__ W2,
                            const float* __restrict__ b2, float* __restrict__ x,
                            const int* __restrict__ lids) {
    int mt = blockIdx.x;
    int wv = threadIdx.x >> 5;
    int lane = threadIdx.x & 31;
    int nn = lane & 15, hl = lane >> 4;
    int b = (mt * 16) / Lp;
    int l0 = (mt * 16) % Lp;
    int n0 = wv * 16;

    v8f c = zero8();
#pragma unroll
    for (int ks = 0; ks < 2; ++ks) {
        int l = l0 + nn;
        v16h a = (l < Lq) ? load_row_f16(Ht + ((size_t)b * Lp + l) * Dm, ks * 32, hl)
                          : zero16();
        v16h bw = load_row_f32cvt(W2 + (size_t)(n0 + nn) * Dm, ks * 32, hl);
        c = WMMA_F16(a, bw, c);
    }
#pragma unroll
    for (int j = 0; j < 8; ++j) {
        int l = l0 + j + 8 * hl;
        if (l < Lq) {
            int n = n0 + nn;
            bool keep = lids[b * Lq + l] != 0;
            size_t idx = ((size_t)b * Lq + l) * Dm + n;
            x[idx] = keep ? (x[idx] + c[j] + b2[n]) : 0.0f;
        }
    }
}

// logits: dot(feats[row], emb[pos[row]]) and neg; one wave per row.
__global__ void logits_kernel(const float* __restrict__ feats, const float* __restrict__ emb,
                              const int* __restrict__ pos, const int* __restrict__ neg,
                              float* __restrict__ outp, float* __restrict__ outn, int nrows) {
    int row = blockIdx.x * 8 + (threadIdx.x >> 5);
    int lane = threadIdx.x & 31;
    if (row >= nrows) return;
    const float* f = feats + (size_t)row * Dm;
    float f0 = f[lane], f1 = f[lane + 32];
    const float* pe = emb + (size_t)pos[row] * Dm;
    const float* ne = emb + (size_t)neg[row] * Dm;
    float dp = f0 * pe[lane] + f1 * pe[lane + 32];
    float dn = f0 * ne[lane] + f1 * ne[lane + 32];
#pragma unroll
    for (int m = 1; m < 32; m <<= 1) {
        dp += __shfl_xor(dp, m, 32);
        dn += __shfl_xor(dn, m, 32);
    }
    if (lane == 0) { outp[row] = dp; outn[row] = dn; }
}

// ---------------- host launcher ----------------
extern "C" void kernel_launch(void* const* d_in, const int* in_sizes, int n_in,
                              void* d_out, int out_size, void* d_ws, size_t ws_size,
                              hipStream_t stream) {
    const int*   log_ids   = (const int*)d_in[0];
    const int*   log_times = (const int*)d_in[1];
    const int*   pos_ids   = (const int*)d_in[2];
    const int*   neg_ids   = (const int*)d_in[3];
    const float* item_emb  = (const float*)d_in[4];
    const float* posK      = (const float*)d_in[5];
    const float* posV      = (const float*)d_in[6];
    const float* tKe       = (const float*)d_in[7];
    const float* tVe       = (const float*)d_in[8];
    const float* ln1g = (const float*)d_in[9],  *ln1b = (const float*)d_in[10];
    const float* Wq  = (const float*)d_in[11], *bq = (const float*)d_in[12];
    const float* Wk  = (const float*)d_in[13], *bk = (const float*)d_in[14];
    const float* Wv  = (const float*)d_in[15], *bv = (const float*)d_in[16];
    const float* ln2g = (const float*)d_in[17], *ln2b = (const float*)d_in[18];
    const float* W1  = (const float*)d_in[19], *b1 = (const float*)d_in[20];
    const float* W2  = (const float*)d_in[21], *b2 = (const float*)d_in[22];
    const float* lnfg = (const float*)d_in[23], *lnfb = (const float*)d_in[24];

    // workspace carve-up (all 16B aligned)
    float* x     = (float*)d_ws;                 // [B,L,64]
    float* qln   = x + (size_t)NROWS * Dm;       // [B,L,64]
    float* attnb = qln + (size_t)NROWS * Dm;     // [B,L,64]
    _Float16* Qh = (_Float16*)(attnb + (size_t)NROWS * Dm);  // [B,Lp,64] f16
    _Float16* Kp = Qh + (size_t)Bc * Lp * Dm;
    _Float16* Vp = Kp + (size_t)Bc * Lp * Dm;
    _Float16* Ht = Vp + (size_t)Bc * Lp * Dm;
    _Float16* QT = Ht + (size_t)Bc * Lp * Dm;    // [B,H,Lp,TPp] f16

    const int MT = Bc * Lp / 16;                 // 448 M-tiles
    const int ATT = Bc * NH * 14;                // 896 attention blocks

    embed_kernel<<<NROWS, Dm, 0, stream>>>(log_ids, item_emb, x);

    for (int i = 0; i < NLAY; ++i) {
        const float* wq = Wq + (size_t)i * Dm * Dm;
        const float* wk = Wk + (size_t)i * Dm * Dm;
        const float* wv = Wv + (size_t)i * Dm * Dm;
        const float* w1 = W1 + (size_t)i * Dm * Dm;
        const float* w2 = W2 + (size_t)i * Dm * Dm;

        ln_kernel<<<NROWS / 8, 256, 0, stream>>>(x, nullptr, qln, ln1g + i * Dm, ln1b + i * Dm, NROWS);
        proj_kernel<<<MT, 128, 0, stream>>>(qln, wq, bq + i * Dm, nullptr, 0, Qh);
        proj_kernel<<<MT, 128, 0, stream>>>(x,   wk, bk + i * Dm, posK,    0, Kp);
        proj_kernel<<<MT, 128, 0, stream>>>(x,   wv, bv + i * Dm, posV,    0, Vp);
        qt_kernel<<<dim3(ATT, TPp / 16), 32, 0, stream>>>(Qh, tKe, QT);
        attn_kernel<<<ATT, 32, 0, stream>>>(Qh, Kp, Vp, QT, log_times, log_ids, tVe, attnb);
        // x = LN2(qln + mha)  (fused residual)
        ln_kernel<<<NROWS / 8, 256, 0, stream>>>(qln, attnb, x, ln2g + i * Dm, ln2b + i * Dm, NROWS);
        proj_kernel<<<MT, 128, 0, stream>>>(x, w1, b1 + i * Dm, nullptr, 1, Ht);
        ffn2_kernel<<<MT, 128, 0, stream>>>(Ht, w2, b2 + i * Dm, x, log_ids);
    }

    ln_kernel<<<NROWS / 8, 256, 0, stream>>>(x, nullptr, qln, lnfg, lnfb, NROWS);
    logits_kernel<<<NROWS / 8, 256, 0, stream>>>(qln, item_emb, pos_ids, neg_ids,
                                                 (float*)d_out, (float*)d_out + NROWS, NROWS);
}
